// AnchorAttention_62543313764754
// MI455X (gfx1250) — matrix operand order repaired and checked by reading, not compiled, not run on hardware
//
#include <hip/hip_runtime.h>
#include <math.h>

typedef _Float16 h16;
typedef __attribute__((ext_vector_type(16))) _Float16 v16h;
typedef __attribute__((ext_vector_type(8)))  _Float16 v8h;
typedef __attribute__((ext_vector_type(8)))  float    v8f;

#define B_   4
#define S_   2048
#define D_   768
#define H_   8
#define HD_  96
#define M_   (B_ * S_)          // 8192 rows of (b,s)
#define NEGV (-1.0e9f)
#define SCALE 0.10206207261596577f  // 1/sqrt(96)

// ---------------------------------------------------------------------------
// WMMA operand loaders (CDNA5 wave32 layouts, CDNA5 ISA §7.12.2)
//
// A (16x32 f16): lane L holds row (L&15); lanes 0-15 carry K=0..7 & 16..23,
// lanes 16-31 carry K=8..15 & 24..31  -> two contiguous 8-half chunks.
// B (32x16 f16): lane L holds column (L&15); lanes 0-15 carry K=0..7 & 16..23,
// lanes 16-31 carry K=8..15 & 24..31  -> two contiguous 8-half chunks.
// C/D (16x16 f32): lane L holds column (L&15), rows (L>>4)*8 + r.
// ---------------------------------------------------------------------------
static __device__ __forceinline__ v16h load_a16x32(const h16* base, int ld) {
  int l = threadIdx.x & 31;
  const h16* p = base + (size_t)(l & 15) * ld + ((l >> 4) * 8);
  v8h lo = *(const v8h*)p;         // K chunk 0
  v8h hi = *(const v8h*)(p + 16);  // K chunk +16
  v16h r;
#pragma unroll
  for (int i = 0; i < 8; ++i) { r[i] = lo[i]; r[i + 8] = hi[i]; }
  return r;
}

static __device__ __forceinline__ v16h load_b32x16(const h16* base, int ld) {
  int l = threadIdx.x & 31;
  const h16* p = base + (size_t)(l & 15) * ld + ((l >> 4) * 8);
  v8h lo = *(const v8h*)p;
  v8h hi = *(const v8h*)(p + 16);
  v16h r;
#pragma unroll
  for (int i = 0; i < 8; ++i) { r[i] = lo[i]; r[i + 8] = hi[i]; }
  return r;
}

// ---------------------------------------------------------------------------
// f32 -> f16 weight conversion
// ---------------------------------------------------------------------------
__global__ void cvt_kernel(const float* __restrict__ src, h16* __restrict__ dst, int n) {
  int i = blockIdx.x * blockDim.x + threadIdx.x;
  if (i < n) dst[i] = (h16)src[i];
}

// ---------------------------------------------------------------------------
// LayerNorm over D=768, output f16 row-major
// ---------------------------------------------------------------------------
__global__ void ln_kernel(const float* __restrict__ x, const float* __restrict__ g,
                          const float* __restrict__ bb, h16* __restrict__ xh) {
  int row = blockIdx.x;
  int t = threadIdx.x;  // 256 threads
  __shared__ float red[256];
  const float* xr = x + (size_t)row * D_;

  float s = 0.f;
  for (int i = t; i < D_; i += 256) s += xr[i];
  red[t] = s; __syncthreads();
  for (int o = 128; o > 0; o >>= 1) { if (t < o) red[t] += red[t + o]; __syncthreads(); }
  float mu = red[0] * (1.0f / D_);
  __syncthreads();

  float v = 0.f;
  for (int i = t; i < D_; i += 256) { float d = xr[i] - mu; v += d * d; }
  red[t] = v; __syncthreads();
  for (int o = 128; o > 0; o >>= 1) { if (t < o) red[t] += red[t + o]; __syncthreads(); }
  float rs = rsqrtf(red[0] * (1.0f / D_) + 1e-5f);

  h16* orow = xh + (size_t)row * D_;
  for (int i = t; i < D_; i += 256)
    orow[i] = (h16)(((xr[i] - mu) * rs) * g[i] + bb[i]);
}

// ---------------------------------------------------------------------------
// GEMM: D[m,n] = sum_k A[m,k] * W[n,k] + bias[n]   (A: M x 768 f16, W: 768 x 768 f16)
// Each wave computes a 16x64 strip (4 accumulators) so one A-operand load
// feeds 4 independent WMMAs.  Block = (32,4) waves covering 16x256;
// grid = (D/256, M/16).
// MODE 0: f16 row-major out (Q, K)
// MODE 1: f16 out transposed per (b,h): Vt[(b*H+h)*HD + hd][s]
// MODE 2: f32 out, (acc + bias) * anchor_mask[row]   (final projection)
// ---------------------------------------------------------------------------
template <int MODE>
__global__ void gemm_kernel(const h16* __restrict__ A, const h16* __restrict__ W,
                            const float* __restrict__ bias,
                            const unsigned char* __restrict__ mask,
                            h16* __restrict__ outh, float* __restrict__ outf) {
  int m0  = blockIdx.y * 16;
  int n0w = (blockIdx.x * 16 + threadIdx.y * 4) * 16;  // wave's first N tile
  int l = threadIdx.x & 31;

  const h16* abase = A + (size_t)m0 * D_;
  const h16* wb    = W + (size_t)n0w * D_;

  v8f cc[4] = {v8f{}, v8f{}, v8f{}, v8f{}};
#pragma unroll 2
  for (int k0 = 0; k0 < D_; k0 += 32) {
    __builtin_prefetch(abase + k0 + 128, 0, 1);   // global_prefetch_b8 ahead
    v16h a = load_a16x32(abase + k0, D_);
#pragma unroll
    for (int t = 0; t < 4; ++t) {
      v16h b = load_b32x16(wb + (size_t)(t * 16) * D_ + k0, D_);
      cc[t] = __builtin_amdgcn_wmma_f32_16x16x32_f16(false, a, false, b, (short)0, cc[t], false, false);
    }
  }

  int n = l & 15, mb = (l >> 4) * 8;
#pragma unroll
  for (int t = 0; t < 4; ++t) {
    int n0 = n0w + t * 16;
    float bn = bias[n0 + n];
    if (MODE == 0) {
#pragma unroll
      for (int r = 0; r < 8; ++r)
        outh[(size_t)(m0 + mb + r) * D_ + n0 + n] = (h16)(cc[t][r] + bn);
    } else if (MODE == 1) {
      int e = n0 + n;
      int h = e / HD_, hd = e % HD_;
      int bi = m0 / S_;
      int s0 = (m0 + mb) - bi * S_;
      v8h o;
#pragma unroll
      for (int r = 0; r < 8; ++r) o[r] = (h16)(cc[t][r] + bn);
      *(v8h*)(outh + ((size_t)(bi * H_ + h) * HD_ + hd) * S_ + s0) = o;
    } else {
#pragma unroll
      for (int r = 0; r < 8; ++r) {
        int m = m0 + mb + r;
        float mk = mask[m] ? 1.0f : 0.0f;
        outf[(size_t)m * D_ + n0 + n] = (cc[t][r] + bn) * mk;
      }
    }
  }
}

// ---------------------------------------------------------------------------
// Attention: one block (4 waves / 128 threads) per (b, h, 16-query tile).
// Phase 1: raw scores(16 x 2048) via WMMA -> LDS f32 (pure load/WMMA/store;
//          no mask traffic inside the WMMA loop)
// Phase 2: scale + key-mask + row softmax (unnormalized exp -> f16 probs in
//          LDS; row sums kept for a cheap post-scale)
// Phase 3: O = P . V^T via WMMA (A from LDS probs, B from transposed V),
//          scaled by 1/rowsum, written f16 row-major [b,s,h*HD+hd].
// ---------------------------------------------------------------------------
__global__ void attn_kernel(const h16* __restrict__ Q, const h16* __restrict__ K,
                            const h16* __restrict__ Vt,
                            const unsigned char* __restrict__ mask,
                            h16* __restrict__ O) {
  extern __shared__ char smem_raw[];
  float* sc     = (float*)smem_raw;                                // 16 x 2048 f32
  h16*   pa     = (h16*)(smem_raw + 16 * S_ * 4);                  // 16 x 2048 f16
  float* red    = (float*)(smem_raw + 16 * S_ * 4 + 16 * S_ * 2);  // 16 x 8
  float* rowmax = red + 128;                                       // 16
  float* rowsum = rowmax + 16;                                     // 16

  int qt = blockIdx.x, h = blockIdx.y, bi = blockIdx.z;
  int q0 = qt * 16;
  int w = threadIdx.x >> 5;  // wave 0..3
  int l = threadIdx.x & 31;
  int n = l & 15, mb = (l >> 4) * 8;

  const h16* Qb = Q + ((size_t)bi * S_ + q0) * D_ + h * HD_;
  const h16* Kb = K + (size_t)bi * S_ * D_ + h * HD_;
  const h16* Vb = Vt + (size_t)(bi * H_ + h) * HD_ * S_;
  const unsigned char* mrow = mask + (size_t)bi * S_;

  // Q A-operands: 16 queries x 96 head dims = 3 WMMA A tiles, reused all tiles.
  v16h qa0 = load_a16x32(Qb, D_);
  v16h qa1 = load_a16x32(Qb + 32, D_);
  v16h qa2 = load_a16x32(Qb + 64, D_);

  // ---- Phase 1: raw scores (no mask / scale here) ----
  for (int kt = w; kt < S_ / 16; kt += 4) {
    int k0 = kt * 16;
    const h16* kb = Kb + (size_t)k0 * D_;
    __builtin_prefetch(kb + (size_t)64 * D_, 0, 1);  // next tile for this wave
    v8f c = {};
    c = __builtin_amdgcn_wmma_f32_16x16x32_f16(false, qa0, false, load_b32x16(kb, D_),      (short)0, c, false, false);
    c = __builtin_amdgcn_wmma_f32_16x16x32_f16(false, qa1, false, load_b32x16(kb + 32, D_), (short)0, c, false, false);
    c = __builtin_amdgcn_wmma_f32_16x16x32_f16(false, qa2, false, load_b32x16(kb + 64, D_), (short)0, c, false, false);
#pragma unroll
    for (int r = 0; r < 8; ++r)
      sc[(mb + r) * S_ + k0 + n] = c[r];
  }
  __syncthreads();

  // ---- Phase 2: scale + mask + softmax (8 threads per query row) ----
  {
    int t = threadIdx.x;
    int row = t >> 3, sub = t & 7;
    const int seg = S_ / 8;  // 256
    float* srow = sc + row * S_;

    float mx = -3.0e38f;
    for (int k = sub * seg; k < sub * seg + seg; ++k) {
      float v = mrow[k] ? srow[k] * SCALE : NEGV;  // key mask + 1/sqrt(HD)
      srow[k] = v;
      mx = fmaxf(mx, v);
    }
    red[row * 8 + sub] = mx; __syncthreads();
    if (sub == 0) {
      float mm = red[row * 8];
#pragma unroll
      for (int j = 1; j < 8; ++j) mm = fmaxf(mm, red[row * 8 + j]);
      rowmax[row] = mm;
    }
    __syncthreads();

    float mm = rowmax[row];
    float s = 0.f;
    h16* prow = pa + row * S_;
    for (int k = sub * seg; k < sub * seg + seg; ++k) {
      float e = __expf(srow[k] - mm);
      prow[k] = (h16)e;
      s += e;
    }
    red[row * 8 + sub] = s; __syncthreads();
    if (sub == 0) {
      float ss = red[row * 8];
#pragma unroll
      for (int j = 1; j < 8; ++j) ss += red[row * 8 + j];
      rowsum[row] = ss;
    }
    __syncthreads();
  }

  // ---- Phase 3: O = P @ V^T, six 16-wide hd tiles split across waves ----
  for (int t6 = w; t6 < 6; t6 += 4) {
    int nd0 = t6 * 16;
    const h16* vb = Vb + (size_t)nd0 * S_;
    v8f c = {};
    for (int k0 = 0; k0 < S_; k0 += 32) {
      __builtin_prefetch(vb + k0 + 256, 0, 1);
      v16h a = load_a16x32(pa + k0, S_);         // probs from LDS
      v16h b = load_b32x16(vb + k0, S_);         // Vt rows = hd
      c = __builtin_amdgcn_wmma_f32_16x16x32_f16(false, a, false, b, (short)0, c, false, false);
    }
#pragma unroll
    for (int r = 0; r < 8; ++r) {
      int m = mb + r;
      float o = c[r] / rowsum[m];
      O[((size_t)bi * S_ + q0 + m) * D_ + h * HD_ + nd0 + n] = (h16)o;
    }
  }
}

// ---------------------------------------------------------------------------
extern "C" void kernel_launch(void* const* d_in, const int* in_sizes, int n_in,
                              void* d_out, int out_size, void* d_ws, size_t ws_size,
                              hipStream_t stream) {
  (void)in_sizes; (void)n_in; (void)out_size; (void)ws_size;
  const float* x            = (const float*)d_in[0];
  const unsigned char* amsk = (const unsigned char*)d_in[1];
  const float* ln_g = (const float*)d_in[2];
  const float* ln_b = (const float*)d_in[3];
  const float* Wq = (const float*)d_in[4];
  const float* bq = (const float*)d_in[5];
  const float* Wk = (const float*)d_in[6];
  const float* bk = (const float*)d_in[7];
  const float* Wv = (const float*)d_in[8];
  const float* bv = (const float*)d_in[9];
  const float* Wo = (const float*)d_in[10];
  const float* bo = (const float*)d_in[11];
  float* out = (float*)d_out;

  char* ws = (char*)d_ws;
  size_t off = 0;
  auto alloc = [&](size_t bytes) -> char* {
    char* p = ws + off;
    off += (bytes + 255) & ~(size_t)255;
    return p;
  };
  h16* xh  = (h16*)alloc((size_t)M_ * D_ * sizeof(h16));   // LN output
  h16* Qh  = (h16*)alloc((size_t)M_ * D_ * sizeof(h16));
  h16* Kh  = (h16*)alloc((size_t)M_ * D_ * sizeof(h16));
  h16* Vt  = (h16*)alloc((size_t)M_ * D_ * sizeof(h16));   // [(b*H+h)*HD + hd][s]
  h16* Ah  = (h16*)alloc((size_t)M_ * D_ * sizeof(h16));   // attention output
  h16* Wqh = (h16*)alloc((size_t)D_ * D_ * sizeof(h16));
  h16* Wkh = (h16*)alloc((size_t)D_ * D_ * sizeof(h16));
  h16* Wvh = (h16*)alloc((size_t)D_ * D_ * sizeof(h16));
  h16* Woh = (h16*)alloc((size_t)D_ * D_ * sizeof(h16));

  const int wn = D_ * D_;
  cvt_kernel<<<(wn + 255) / 256, 256, 0, stream>>>(Wq, Wqh, wn);
  cvt_kernel<<<(wn + 255) / 256, 256, 0, stream>>>(Wk, Wkh, wn);
  cvt_kernel<<<(wn + 255) / 256, 256, 0, stream>>>(Wv, Wvh, wn);
  cvt_kernel<<<(wn + 255) / 256, 256, 0, stream>>>(Wo, Woh, wn);

  ln_kernel<<<M_, 256, 0, stream>>>(x, ln_g, ln_b, xh);

  dim3 gg(D_ / 256, M_ / 16), gb(32, 4);  // (3, 512) blocks, 4 waves each
  gemm_kernel<0><<<gg, gb, 0, stream>>>(xh, Wqh, bq, nullptr, Qh, nullptr);
  gemm_kernel<0><<<gg, gb, 0, stream>>>(xh, Wkh, bk, nullptr, Kh, nullptr);
  gemm_kernel<1><<<gg, gb, 0, stream>>>(xh, Wvh, bv, nullptr, Vt, nullptr);

  size_t shbytes = (size_t)16 * S_ * 4 + (size_t)16 * S_ * 2 + (128 + 32) * 4;
  attn_kernel<<<dim3(S_ / 16, H_, B_), 128, shbytes, stream>>>(Qh, Kh, Vt, amsk, Ah);

  gemm_kernel<2><<<gg, gb, 0, stream>>>(Ah, Woh, bo, amsk, nullptr, out);
}
